// GnnModel_26018911879251
// MI455X (gfx1250) — matrix operand order repaired and checked by reading, not compiled, not run on hardware
//
#include <hip/hip_runtime.h>
#include <hip/hip_bf16.h>

// ---------------- model constants ----------------
#define NN      131072      // nodes
#define NE      524288      // edges
#define NB      4096        // batch
#define FIN     70
#define HIDD    128
#define OUTD    64
#define NCELLS  4
#define RELD    16384       // HID*HID
#define NLBL    86

typedef __attribute__((ext_vector_type(16))) _Float16 v16h;
typedef __attribute__((ext_vector_type(8)))  float    v8f;

__device__ __forceinline__ float wred_sum(float v) {
#pragma unroll
  for (int o = 16; o > 0; o >>= 1) v += __shfl_xor(v, o, 32);
  return v;
}
__device__ __forceinline__ float eluf(float x) { return x > 0.f ? x : (expf(x) - 1.f); }

// ---- layernorm over F_IN dims, output into 128-stride buffer (cols >= FIN left untouched)
__global__ void k_ln_fin(const float* __restrict__ x, const float* __restrict__ g,
                         const float* __restrict__ b, float* __restrict__ out) {
  int wave = threadIdx.x >> 5, lane = threadIdx.x & 31;
  int row = blockIdx.x * 8 + wave;
  if (row >= NN) return;
  const float* xr = x + (size_t)row * FIN;
  float v0 = (lane      < FIN) ? xr[lane]      : 0.f;
  float v1 = (lane + 32 < FIN) ? xr[lane + 32] : 0.f;
  float v2 = (lane + 64 < FIN) ? xr[lane + 64] : 0.f;
  float s  = wred_sum(v0 + v1 + v2);
  float ss = wred_sum(v0 * v0 + v1 * v1 + v2 * v2);
  float mean = s / (float)FIN;
  float var  = ss / (float)FIN - mean * mean;
  float rstd = rsqrtf(var + 1e-5f);
  float* orow = out + (size_t)row * HIDD;
  if (lane      < FIN) orow[lane]      = (v0 - mean) * rstd * g[lane]      + b[lane];
  if (lane + 32 < FIN) orow[lane + 32] = (v1 - mean) * rstd * g[lane + 32] + b[lane + 32];
  if (lane + 64 < FIN) orow[lane + 64] = (v2 - mean) * rstd * g[lane + 64] + b[lane + 64];
}

// ---- layernorm over 128 dims + elu (feeds next cell)
__global__ void k_ln_elu(const float* __restrict__ x, const float* __restrict__ g,
                         const float* __restrict__ b, float* __restrict__ out) {
  int wave = threadIdx.x >> 5, lane = threadIdx.x & 31;
  int n = blockIdx.x * 8 + wave;
  if (n >= NN) return;
  const float* r = x + (size_t)n * HIDD;
  float v[4], s = 0.f, ss = 0.f;
#pragma unroll
  for (int j = 0; j < 4; j++) { v[j] = r[lane + 32 * j]; s += v[j]; ss += v[j] * v[j]; }
  s = wred_sum(s); ss = wred_sum(ss);
  float mean = s * (1.f / 128.f);
  float var  = ss * (1.f / 128.f) - mean * mean;
  float rstd = rsqrtf(var + 1e-5f);
  float* orow = out + (size_t)n * HIDD;
#pragma unroll
  for (int j = 0; j < 4; j++) {
    int c = lane + 32 * j;
    orow[c] = eluf((v[j] - mean) * rstd * g[c] + b[c]);
  }
}

// ---- elu + convert to f16 (2 cols/thread, dword store), zero-pad cols >= din
__global__ void k_elu_h(const float* __restrict__ x, _Float16* __restrict__ out, int din) {
  int tid = blockIdx.x * 256 + threadIdx.x;      // NN*64 threads
  int n = tid >> 6, c2 = (tid & 63) * 2;
  union { _Float16 h[2]; unsigned u; } p;
  float v0 = (c2     < din) ? eluf(x[(size_t)n * HIDD + c2])     : 0.f;
  float v1 = (c2 + 1 < din) ? eluf(x[(size_t)n * HIDD + c2 + 1]) : 0.f;
  p.h[0] = (_Float16)v0; p.h[1] = (_Float16)v1;
  *(unsigned*)(out + (size_t)n * HIDD + c2) = p.u;
}

// ---- pack weight W[din x 128] (row-major) -> Wt f16 [128 x 128], Wt[n][k]=W[k][n], k>=din -> 0
__global__ void k_pack_w(const float* __restrict__ W, _Float16* __restrict__ wt, int din) {
  int tid = blockIdx.x * 256 + threadIdx.x;   // 16384 threads
  int n = tid >> 7, k = tid & 127;
  wt[(size_t)n * HIDD + k] = (k < din) ? (_Float16)W[(size_t)k * HIDD + n] : (_Float16)0.f;
}

// ---- WMMA GEMM: C[NN x 128] = A_f16[NN x 128] * Wt^T  (Wt stored [n][k])
// block: 256 threads = 8 waves; wave w -> cols [16w,16w+16); block -> 64 rows (4 M-tiles/wave).
// Per K-step: stage B + all 4 A fragments (10 b128 loads in flight), then 4 back-to-back WMMAs.
__global__ void k_gemm(const _Float16* __restrict__ A,
                       const _Float16* __restrict__ Bt,
                       float* __restrict__ C) {
  int wave = threadIdx.x >> 5, lane = threadIdx.x & 31;
  int half = lane >> 4, l15 = lane & 15;
  int row0 = blockIdx.x << 6;                 // 64 rows per block
  int col  = (wave << 4) + l15;
  const _Float16* brow = Bt + (size_t)col * HIDD;
  const _Float16* arow0 = A + (size_t)(row0 + l15) * HIDD;
  // warm WGP cache for this lane's A rows (global_prefetch_b8 path)
  __builtin_prefetch(arow0, 0, 1);
  __builtin_prefetch(arow0 + (size_t)32 * HIDD, 0, 1);
  v8f acc[4] = {};
  union U { v16h v; uint4 u[2]; };
#pragma unroll
  for (int k0 = 0; k0 < HIDD; k0 += 32) {
    // B 32x16 fragment: lanes 0..15 K 0..15, lanes 16..31 K 16..31; n = lane&15
    U b;
    b.u[0] = *(const uint4*)(brow + k0 + half * 16);
    b.u[1] = *(const uint4*)(brow + k0 + half * 16 + 8);
    // stage all 4 A fragments first so their loads overlap
    U a[4];
#pragma unroll
    for (int mt = 0; mt < 4; mt++) {
      const _Float16* arow = arow0 + (size_t)(mt * 16) * HIDD;
      // A 16x32 f16 fragment: lane<16 -> K 0..15; lane>=16 -> K 16..31
      a[mt].u[0] = *(const uint4*)(arow + k0 + half * 16);
      a[mt].u[1] = *(const uint4*)(arow + k0 + half * 16 + 8);
    }
#pragma unroll
    for (int mt = 0; mt < 4; mt++) {
      acc[mt] = __builtin_amdgcn_wmma_f32_16x16x32_f16(false, a[mt].v, false, b.v,
                                                       (short)0, acc[mt], false, false);
    }
  }
#pragma unroll
  for (int mt = 0; mt < 4; mt++) {
#pragma unroll
    for (int v = 0; v < 8; v++) {
      int r = row0 + mt * 16 + v + half * 8;  // C/D layout: lane<16 M=v, lane>=16 M=v+8
      C[(size_t)r * HIDD + col] = acc[mt][v];
    }
  }
}

// ---- per-node attention logits: l[n][h] = sum_c hs[n][h*64+c]*a[h*64+c]
__global__ void k_dotl(const float* __restrict__ hs, const float* __restrict__ a,
                       float* __restrict__ l) {
  int wave = threadIdx.x >> 5, lane = threadIdx.x & 31;
  int n = blockIdx.x * 8 + wave;
  if (n >= NN) return;
  const float* r = hs + (size_t)n * HIDD;
  float p0 = r[lane] * a[lane] + r[lane + 32] * a[lane + 32];
  float p1 = r[lane + 64] * a[lane + 64] + r[lane + 96] * a[lane + 96];
  p0 = wred_sum(p0); p1 = wred_sum(p1);
  if (lane == 0) { l[n * 2] = p0; l[n * 2 + 1] = p1; }
}

__global__ void k_fill(float* __restrict__ p, float v, int n) {
  int t = blockIdx.x * 256 + threadIdx.x;
  if (t < n) p[t] = v;
}

// ---- edge pass 1: leaky score + segment max (float atomic max via int/uint trick)
__global__ void k_edge1(const int* __restrict__ src, const int* __restrict__ dst,
                        const float* __restrict__ ls, const float* __restrict__ ld,
                        float* __restrict__ ew, float* __restrict__ m) {
  int t = blockIdx.x * 256 + threadIdx.x;   // 2*NE threads
  int e = t >> 1, h = t & 1;
  int s = src[e], d = dst[e];
  float v = ls[s * 2 + h] + ld[d * 2 + h];
  v = v > 0.f ? v : 0.2f * v;
  ew[t] = v;
  int* ai = (int*)&m[d * 2 + h];
  if (v >= 0.f) atomicMax(ai, __float_as_int(v));
  else          atomicMin((unsigned int*)ai, (unsigned int)__float_as_int(v));
}

// ---- edge pass 2: w = exp(e - max); segment sum
__global__ void k_edge2(const int* __restrict__ dst, float* __restrict__ ew,
                        const float* __restrict__ m, float* __restrict__ z) {
  int t = blockIdx.x * 256 + threadIdx.x;
  int e = t >> 1, h = t & 1;
  int d = dst[e];
  float w = expf(ew[t] - m[d * 2 + h]);
  ew[t] = w;
  atomicAdd(&z[d * 2 + h], w);
}

// ---- edge pass 3: weighted feature scatter (both heads summed; /2 deferred)
__global__ __launch_bounds__(256) void k_edge3(const int* __restrict__ src,
                                               const int* __restrict__ dst,
                                               const float* __restrict__ hs,
                                               const float* __restrict__ ew,
                                               const float* __restrict__ z,
                                               float* __restrict__ out, int off) {
  int t = threadIdx.x;
  int e = blockIdx.x * 4 + (t >> 6);
  int c = t & 63;
  int s = src[e], d = dst[e];
  float a0 = ew[e * 2]     / (z[d * 2]     + 1e-16f);
  float a1 = ew[e * 2 + 1] / (z[d * 2 + 1] + 1e-16f);
  float v = a0 * hs[(size_t)s * HIDD + c] + a1 * hs[(size_t)s * HIDD + 64 + c];
  atomicAdd(&out[(size_t)d * HIDD + off + c], v);
}

// ---- apply mean over heads (x0.5) + bias, in place on one 64-wide half
__global__ void k_finish(float* __restrict__ out, const float* __restrict__ bias, int off) {
  int t = blockIdx.x * 256 + threadIdx.x;   // NN*64 threads
  int n = t >> 6, c = t & 63;
  float* p = &out[(size_t)n * HIDD + off + c];
  *p = *p * 0.5f + bias[c];
}

// ---- tanh(x.w_score) gated pooling into batch embeddings
__global__ void k_pool(const float* __restrict__ x, const float* __restrict__ wsc,
                       const int* __restrict__ batch, float* __restrict__ rep) {
  int wave = threadIdx.x >> 5, lane = threadIdx.x & 31;
  int n = blockIdx.x * 8 + wave;
  if (n >= NN) return;
  const float* r = x + (size_t)n * HIDD;
  float dacc = 0.f;
#pragma unroll
  for (int j = 0; j < 4; j++) dacc += r[lane + 32 * j] * wsc[lane + 32 * j];
  dacc = wred_sum(dacc);
  float s = tanhf(dacc);
  int b = batch[n];
#pragma unroll
  for (int j = 0; j < 4; j++) {
    int c = lane + 32 * j;
    atomicAdd(&rep[(size_t)b * HIDD + c], r[c] * s);
  }
}

// ---- l2-normalize 128-length rows in place
__global__ void k_nrow(float* __restrict__ x, int rows) {
  int wave = threadIdx.x >> 5, lane = threadIdx.x & 31;
  int row = blockIdx.x * 8 + wave;
  if (row >= rows) return;
  float* r = x + (size_t)row * HIDD;
  float v[4], ss = 0.f;
#pragma unroll
  for (int j = 0; j < 4; j++) { v[j] = r[lane + 32 * j]; ss += v[j] * v[j]; }
  ss = wred_sum(ss);
  float inv = 1.f / fmaxf(sqrtf(ss), 1e-12f);
#pragma unroll
  for (int j = 0; j < 4; j++) r[lane + 32 * j] = v[j] * inv;
}

// ---- bilinear score: S[b][c][d] = (Hn[c] . R . Tn[d]) / ||R_row||
__global__ __launch_bounds__(256) void k_score(const float* __restrict__ repsH,
                                               const float* __restrict__ repsT,
                                               const float* __restrict__ relE,
                                               const int* __restrict__ rels,
                                               float* __restrict__ S) {
  __shared__ float sH[4 * HIDD], sT[4 * HIDD], sP[HIDD * 4], sr[8];
  int b = blockIdx.x, t = threadIdx.x;
  for (int i = t; i < 4 * HIDD; i += 256) {
    int cell = i >> 7, c = i & 127;
    sH[i] = repsH[((size_t)cell * NB + b) * HIDD + c];
    sT[i] = repsT[((size_t)cell * NB + b) * HIDD + c];
  }
  __syncthreads();
  const float* R = relE + (size_t)rels[b] * RELD;
  float ss = 0.f;
  if (t < HIDD) {
    float a0 = 0.f, a1 = 0.f, a2 = 0.f, a3 = 0.f;
    const float* Rr = R + (size_t)t * HIDD;
    for (int k = 0; k < HIDD; k++) {
      float rv = Rr[k];
      ss += rv * rv;
      a0 += rv * sT[k]; a1 += rv * sT[HIDD + k];
      a2 += rv * sT[2 * HIDD + k]; a3 += rv * sT[3 * HIDD + k];
    }
    sP[t * 4 + 0] = a0; sP[t * 4 + 1] = a1; sP[t * 4 + 2] = a2; sP[t * 4 + 3] = a3;
  }
  ss = wred_sum(ss);
  if ((t & 31) == 0 && t < HIDD) sr[t >> 5] = ss;
  __syncthreads();
  if (t == 0) sr[0] = sr[0] + sr[1] + sr[2] + sr[3];
  __syncthreads();
  float inv = 1.f / fmaxf(sqrtf(sr[0]), 1e-12f);
  if (t < 16) {
    int c = t >> 2, d = t & 3;
    float acc = 0.f;
    for (int h = 0; h < HIDD; h++) acc += sH[c * HIDD + h] * sP[h * 4 + d];
    S[(size_t)b * 16 + t] = acc * inv;
  }
}

// ---- prelu + tiny MLP
__global__ void k_mlp(const float* __restrict__ S, const float* __restrict__ W,
                      const float* __restrict__ bias, const float* __restrict__ pa,
                      float* __restrict__ out) {
  int b = blockIdx.x, l = threadIdx.x;
  if (l >= NLBL) return;
  float a = pa[0];
  float acc = bias[l];
  const float* s = S + (size_t)b * 16;
#pragma unroll
  for (int j = 0; j < 16; j++) {
    float v = s[j]; v = v >= 0.f ? v : a * v;
    acc += v * W[j * NLBL + l];
  }
  out[(size_t)b * NLBL + l] = acc;
}

// ---------------- host orchestration ----------------
struct CellP {
  const float *W, *Wb_s, *Wb_d, *a_s, *a_d, *ab_s, *ab_d,
              *b_conv, *b_bi, *w_score, *norm_g, *norm_b;
};

extern "C" void kernel_launch(void* const* d_in, const int* in_sizes, int n_in,
                              void* d_out, int out_size, void* d_ws, size_t ws_size,
                              hipStream_t stream) {
  // ---- resolve input layout (two plausible pytree flattenings) ----
  const float *h_x, *t_x;
  const int *h_ei, *t_ei, *b_ei, *h_batch, *t_batch, *rels;
  int pb;
  if (in_sizes[0] == NN * FIN) {                 // top-level dict insertion order
    h_x = (const float*)d_in[0]; t_x = (const float*)d_in[1];
    h_ei = (const int*)d_in[2]; t_ei = (const int*)d_in[3]; b_ei = (const int*)d_in[4];
    h_batch = (const int*)d_in[5]; t_batch = (const int*)d_in[6]; rels = (const int*)d_in[7];
    pb = 8;
  } else {                                       // fully sorted tree_leaves order
    b_ei = (const int*)d_in[0]; h_batch = (const int*)d_in[1]; h_ei = (const int*)d_in[2];
    h_x = (const float*)d_in[3]; pb = 4;
    rels = (const int*)d_in[n_in - 4]; t_batch = (const int*)d_in[n_in - 3];
    t_ei = (const int*)d_in[n_in - 2]; t_x = (const float*)d_in[n_in - 1];
  }
  auto F = [&](int i) { return (const float*)d_in[i]; };
  CellP cp[NCELLS];
  const float *ing, *inb, *mlpW, *mlpb, *prelu, *relE;
  if (in_sizes[pb] == FIN * HIDD) {
    // sorted params pytree: cells first; per-cell key order:
    // W, Wb_d, Wb_s, a_d, a_s, ab_d, ab_s, b_bi, b_conv, norm_b, norm_g, w_score
    for (int i = 0; i < NCELLS; i++) {
      int q = pb + i * 12;
      cp[i].W = F(q + 0);  cp[i].Wb_d = F(q + 1); cp[i].Wb_s = F(q + 2);
      cp[i].a_d = F(q + 3); cp[i].a_s = F(q + 4);
      cp[i].ab_d = F(q + 5); cp[i].ab_s = F(q + 6);
      cp[i].b_bi = F(q + 7); cp[i].b_conv = F(q + 8);
      cp[i].norm_b = F(q + 9); cp[i].norm_g = F(q + 10); cp[i].w_score = F(q + 11);
    }
    inb = F(pb + 48); ing = F(pb + 49);
    mlpW = F(pb + 50); mlpb = F(pb + 51); prelu = F(pb + 52); relE = F(pb + 53);
  } else {
    // insertion-order params: init_norm{g,b}, cells[W,a_s,a_d,b_conv,Wb_s,Wb_d,ab_s,ab_d,
    //                          b_bi,w_score,norm_g,norm_b], rel_emb, prelu, mlp_W, mlp_b
    ing = F(pb); inb = F(pb + 1);
    for (int i = 0; i < NCELLS; i++) {
      int q = pb + 2 + i * 12;
      cp[i].W = F(q + 0); cp[i].a_s = F(q + 1); cp[i].a_d = F(q + 2); cp[i].b_conv = F(q + 3);
      cp[i].Wb_s = F(q + 4); cp[i].Wb_d = F(q + 5);
      cp[i].ab_s = F(q + 6); cp[i].ab_d = F(q + 7); cp[i].b_bi = F(q + 8);
      cp[i].w_score = F(q + 9); cp[i].norm_g = F(q + 10); cp[i].norm_b = F(q + 11);
    }
    relE = F(pb + 50); prelu = F(pb + 51); mlpW = F(pb + 52); mlpb = F(pb + 53);
  }

  // ---- workspace layout (bytes) ----
  const size_t SZ_NH = (size_t)NN * HIDD * 4;            // 64 MiB
  size_t off = 0;
  auto take = [&](size_t bytes) { size_t r = off; off = (off + bytes + 255) & ~(size_t)255; return r; };
  char* ws = (char*)d_ws;
  float*    hx   = (float*)(ws + take(SZ_NH));
  float*    tx   = (float*)(ws + take(SZ_NH));
  float*    hnew = (float*)(ws + take(SZ_NH));
  float*    tnew = (float*)(ws + take(SZ_NH));
  float*    g0   = (float*)(ws + take(SZ_NH));
  float*    g1   = (float*)(ws + take(SZ_NH));
  _Float16* abf  = (_Float16*)(ws + take((size_t)NN * HIDD * 2));
  _Float16* tbf  = (_Float16*)(ws + take((size_t)NN * HIDD * 2));
  _Float16* wpk  = (_Float16*)(ws + take((size_t)NCELLS * 3 * HIDD * HIDD * 2));
  float*    lsb  = (float*)(ws + take((size_t)NN * 2 * 4));
  float*    ldb  = (float*)(ws + take((size_t)NN * 2 * 4));
  float*    mb   = (float*)(ws + take((size_t)NN * 2 * 4));
  float*    zb   = (float*)(ws + take((size_t)NN * 2 * 4));
  float*    ewb  = (float*)(ws + take((size_t)NE * 2 * 4));
  float*    repsH= (float*)(ws + take((size_t)NCELLS * NB * HIDD * 4));
  float*    repsT= (float*)(ws + take((size_t)NCELLS * NB * HIDD * 4));
  float*    Sbuf = (float*)(ws + take((size_t)NB * 16 * 4));
  if (ws_size < off) return;   // not enough scratch; bail deterministically

  // ---- pack all weights to f16 transposed (12 x [128x128]) ----
  for (int i = 0; i < NCELLS; i++) {
    int din = (i == 0) ? FIN : HIDD;
    k_pack_w<<<64, 256, 0, stream>>>(cp[i].W,    wpk + (size_t)(i * 3 + 0) * HIDD * HIDD, din);
    k_pack_w<<<64, 256, 0, stream>>>(cp[i].Wb_s, wpk + (size_t)(i * 3 + 1) * HIDD * HIDD, din);
    k_pack_w<<<64, 256, 0, stream>>>(cp[i].Wb_d, wpk + (size_t)(i * 3 + 2) * HIDD * HIDD, din);
  }

  // ---- initial layernorm (70 dims) ----
  k_ln_fin<<<NN / 8, 256, 0, stream>>>(h_x, ing, inb, hx);
  k_ln_fin<<<NN / 8, 256, 0, stream>>>(t_x, ing, inb, tx);
  k_fill<<<(NCELLS * NB * HIDD) / 256, 256, 0, stream>>>(repsH, 0.f, NCELLS * NB * HIDD);
  k_fill<<<(NCELLS * NB * HIDD) / 256, 256, 0, stream>>>(repsT, 0.f, NCELLS * NB * HIDD);

  auto gat = [&](const float* ghs, const float* ghd, const float* as_, const float* ad_,
                 const int* src, const int* dst, float* outb, int offc, const float* bias) {
    k_dotl<<<NN / 8, 256, 0, stream>>>(ghs, as_, lsb);
    k_dotl<<<NN / 8, 256, 0, stream>>>(ghd, ad_, ldb);
    k_fill<<<(NN * 2) / 256, 256, 0, stream>>>(mb, -INFINITY, NN * 2);
    k_fill<<<(NN * 2) / 256, 256, 0, stream>>>(zb, 0.f, NN * 2);
    k_edge1<<<(2 * NE) / 256, 256, 0, stream>>>(src, dst, lsb, ldb, ewb, mb);
    k_edge2<<<(2 * NE) / 256, 256, 0, stream>>>(dst, ewb, mb, zb);
    k_edge3<<<NE / 4, 256, 0, stream>>>(src, dst, ghs, ewb, zb, outb, offc);
    k_finish<<<(NN * 64) / 256, 256, 0, stream>>>(outb, bias, offc);
  };

  for (int i = 0; i < NCELLS; i++) {
    const CellP& C = cp[i];
    int din = (i == 0) ? FIN : HIDD;
    const _Float16* wW  = wpk + (size_t)(i * 3 + 0) * HIDD * HIDD;
    const _Float16* wBs = wpk + (size_t)(i * 3 + 1) * HIDD * HIDD;
    const _Float16* wBd = wpk + (size_t)(i * 3 + 2) * HIDD * HIDD;

    k_elu_h<<<(NN * 64) / 256, 256, 0, stream>>>(hx, abf, din);   // ah (f16, padded)
    k_elu_h<<<(NN * 64) / 256, 256, 0, stream>>>(tx, tbf, din);   // at
    k_fill<<<(NN * HIDD) / 256, 256, 0, stream>>>(hnew, 0.f, NN * HIDD);
    k_fill<<<(NN * HIDD) / 256, 256, 0, stream>>>(tnew, 0.f, NN * HIDD);

    // h_rep: hs = hd = ah@W ; edges h_ei ; -> h_new[:, 0:64]
    k_gemm<<<NN / 64, 256, 0, stream>>>(abf, wW, g0);
    gat(g0, g0, C.a_s, C.a_d, h_ei, h_ei + NE, hnew, 0, C.b_conv);

    // t_rep: at@W ; edges t_ei ; -> t_new[:, 0:64]
    k_gemm<<<NN / 64, 256, 0, stream>>>(tbf, wW, g0);
    gat(g0, g0, C.a_s, C.a_d, t_ei, t_ei + NE, tnew, 0, C.b_conv);

    // t_bi: hs = ah@Wb_s, hd = at@Wb_d ; src=b_ei[0], dst=b_ei[1] ; -> t_new[:, 64:]
    k_gemm<<<NN / 64, 256, 0, stream>>>(abf, wBs, g0);
    k_gemm<<<NN / 64, 256, 0, stream>>>(tbf, wBd, g1);
    gat(g0, g1, C.ab_s, C.ab_d, b_ei, b_ei + NE, tnew, 64, C.b_bi);

    // h_bi: hs = at@Wb_s, hd = ah@Wb_d ; reversed edges ; -> h_new[:, 64:]
    k_gemm<<<NN / 64, 256, 0, stream>>>(tbf, wBs, g0);
    k_gemm<<<NN / 64, 256, 0, stream>>>(abf, wBd, g1);
    gat(g0, g1, C.ab_s, C.ab_d, b_ei + NE, b_ei, hnew, 64, C.b_bi);

    // gated pooling into batch embeddings for this cell
    k_pool<<<NN / 8, 256, 0, stream>>>(hnew, C.w_score, h_batch, repsH + (size_t)i * NB * HIDD);
    k_pool<<<NN / 8, 256, 0, stream>>>(tnew, C.w_score, t_batch, repsT + (size_t)i * NB * HIDD);

    // next-cell inputs: elu(layer_norm(x_new))
    k_ln_elu<<<NN / 8, 256, 0, stream>>>(hnew, C.norm_g, C.norm_b, hx);
    k_ln_elu<<<NN / 8, 256, 0, stream>>>(tnew, C.norm_g, C.norm_b, tx);
  }

  // ---- final scoring ----
  k_nrow<<<(NCELLS * NB) / 8, 256, 0, stream>>>(repsH, NCELLS * NB);
  k_nrow<<<(NCELLS * NB) / 8, 256, 0, stream>>>(repsT, NCELLS * NB);
  k_score<<<NB, 256, 0, stream>>>(repsH, repsT, relE, rels, Sbuf);
  k_mlp<<<NB, 96, 0, stream>>>(Sbuf, mlpW, mlpb, prelu, (float*)d_out);
}
